// ConvAttention_35613868818762
// MI455X (gfx1250) — matrix operand order repaired and checked by reading, not compiled, not run on hardware
//
#include <hip/hip_runtime.h>

typedef __bf16 bf16_t;
typedef __attribute__((ext_vector_type(16))) __bf16 v16bf;
typedef __attribute__((ext_vector_type(8)))  float  v8f;

union ABFrag { v16bf bf; unsigned int u[8]; };

__device__ __forceinline__ unsigned short f2bf(float f) {
    unsigned int x = __float_as_uint(f);
    x += 0x7FFFu + ((x >> 16) & 1u);   // round-to-nearest-even
    return (unsigned short)(x >> 16);
}

#define WMMA_BF16(A, B, C) \
    __builtin_amdgcn_wmma_f32_16x16x32_bf16(false, (A), false, (B), (short)0, (C), false, false)

// gfx1250 async DMA: global -> LDS, 16B per lane, tracked by ASYNCcnt.
__device__ __forceinline__ void async_copy_b128(unsigned lds_byte_addr, const void* gaddr) {
    asm volatile("global_load_async_to_lds_b128 %0, %1, off"
                 :: "v"(lds_byte_addr), "v"(gaddr) : "memory");
}
__device__ __forceinline__ void wait_async0() {
#if __has_builtin(__builtin_amdgcn_s_wait_asynccnt)
    __builtin_amdgcn_s_wait_asynccnt(0);
#else
    asm volatile("s_wait_asynccnt 0x0" ::: "memory");
#endif
}

// ---------------------------------------------------------------------------
// Kernel 1: w_out (f32, [o][k] 256x256) -> bf16 transposed [k][o] (B operand)
// ---------------------------------------------------------------------------
__global__ void prep_w_kernel(const float* __restrict__ w, unsigned short* __restrict__ wbf) {
    int i = blockIdx.x * blockDim.x + threadIdx.x;   // 65536 exact
    int o = i >> 8, k = i & 255;
    wbf[k * 256 + o] = f2bf(w[i]);
}

// ---------------------------------------------------------------------------
// Kernel 2: fused depthwise conv1 (k=5 along H, pad 2) + conv2 (k=8 stride 8
// along W). One thread per output (b, oc, h, wq). Each block covers 8 h x
// 32 wq of one channel, so k-outputs are transposed through LDS and written
// as b128 stores instead of scattered 2B stores.
// ---------------------------------------------------------------------------
__global__ void __launch_bounds__(256) conv_qkv_kernel(
    const float* __restrict__ x, const float* __restrict__ w1,
    const float* __restrict__ w2,
    unsigned short* __restrict__ qb, unsigned short* __restrict__ kTb,
    unsigned short* __restrict__ vb)
{
    const int DIM = 16, H = 128, W = 256, OC = 384, WQ = 32, C = 128;
    __shared__ unsigned short ktile[8 * 32];         // 512 B transpose staging

    int lt  = threadIdx.x;
    int tid = blockIdx.x * 256 + lt;                 // 50,331,648 exact
    int wq = tid & 31;
    int h  = (tid >> 5) & 127;
    int oc = (tid >> 12) % OC;
    int b  = tid / (4096 * OC);
    int g  = oc / 24;                                // groups=16, 24 oc per group

    float w1v[5], w2v[8];
#pragma unroll
    for (int t = 0; t < 5; ++t) w1v[t] = w1[oc * 5 + t];
#pragma unroll
    for (int j = 0; j < 8; ++j) w2v[j] = w2[oc * 8 + j];

    const float* xbase = x + ((b * DIM + g) * H) * W + wq * 8;
    float acc = 0.f;
#pragma unroll
    for (int t = 0; t < 5; ++t) {
        int hh = h + t - 2;
        if (hh < 0 || hh >= H) continue;
        const float* xr = xbase + hh * W;
        float s = 0.f;
#pragma unroll
        for (int j = 0; j < 8; ++j) s += w2v[j] * xr[j];
        acc += w1v[t] * s;
    }

    int sel = oc >> 7;        // 0=q 1=k 2=v (uniform per block)
    int c   = oc & 127;
    if (sel == 0) {
        qb[((b * C + c) * H + h) * WQ + wq] = f2bf(acc * 0.17677669529663687f);
    } else if (sel == 1) {
        // transpose 8h x 32wq tile -> kT[d][seq], b128 stores
        ktile[(lt >> 5) * 32 + (lt & 31)] = f2bf(acc);
        __syncthreads();
        if (lt < 32) {
            int wq2 = lt;
            int h0  = h;      // hl==0 for lt<32, so h == block's h base (multiple of 8)
            unsigned int p0 = (unsigned)ktile[0 * 32 + wq2] | ((unsigned)ktile[1 * 32 + wq2] << 16);
            unsigned int p1 = (unsigned)ktile[2 * 32 + wq2] | ((unsigned)ktile[3 * 32 + wq2] << 16);
            unsigned int p2 = (unsigned)ktile[4 * 32 + wq2] | ((unsigned)ktile[5 * 32 + wq2] << 16);
            unsigned int p3 = (unsigned)ktile[6 * 32 + wq2] | ((unsigned)ktile[7 * 32 + wq2] << 16);
            uint4 val = {p0, p1, p2, p3};
            *(uint4*)(kTb + ((b * C + c) * WQ + wq2) * H + h0) = val;
        }
    } else {
        vb[((b * C + c) * H + h) * WQ + wq] = f2bf(acc);
    }
}

// ---------------------------------------------------------------------------
// Kernel 3: attention. One block (8 waves) per head (b,c). seq=128, d=32.
// q/kT/v staged via async global->LDS DMA (24 KB). Wave w owns M-tile rows
// [16w,16w+16): scores (1 WMMA per 16x16 tile, K=32), shfl_xor softmax,
// attn->bf16->LDS, then attn@v. Output written pre-permuted.
// ---------------------------------------------------------------------------
__global__ void __launch_bounds__(256) attn_kernel(
    const unsigned short* __restrict__ qb, const unsigned short* __restrict__ kTb,
    const unsigned short* __restrict__ vb, unsigned short* __restrict__ pre)
{
    __shared__ unsigned short smem[3 * 4096 + 16384];   // q | kT | v | attn (56 KB)
    unsigned short* qS = smem;          // [seq][d]
    unsigned short* kS = smem + 4096;   // [d][seq]
    unsigned short* vS = smem + 8192;   // [seq][d]
    unsigned short* aS = smem + 12288;  // attn bf16 [128][128]

    int head = blockIdx.x;            // b*128 + c
    int b = head >> 7, c = head & 127;
    int tid = threadIdx.x;

    // --- async DMA: 3 x 8KB contiguous in LDS = 1536 b128 chunks, 6/thread ---
    unsigned lds_base = (unsigned)(size_t)&smem[0];
    const char* gq = (const char*)(qb  + (size_t)head * 4096);
    const char* gk = (const char*)(kTb + (size_t)head * 4096);
    const char* gv = (const char*)(vb  + (size_t)head * 4096);
#pragma unroll
    for (int k2 = 0; k2 < 6; ++k2) {
        int cid = tid + k2 * 256;
        int buf = cid >> 9;                      // 512 chunks per 8KB buffer
        int off = (cid & 511) << 4;
        const char* src = (buf == 0) ? gq + off : (buf == 1) ? gk + off : gv + off;
        async_copy_b128(lds_base + (unsigned)(cid << 4), src);
    }
    wait_async0();
    __syncthreads();

    int wave = tid >> 5;
    int lane = tid & 31;
    int mloc = lane & 15;
    int half = lane >> 4;
    int m    = wave * 16 + mloc;       // A-operand row owned by this lane

    unsigned int* qS32 = (unsigned int*)qS;
    unsigned int* kS32 = (unsigned int*)kS;
    unsigned int* vS32 = (unsigned int*)vS;

    ABFrag af, bf;
    // --- scores = q @ k^T (per tile: single 16x16x32 WMMA) ---
#pragma unroll
    for (int g2 = 0; g2 < 2; ++g2)
#pragma unroll
        for (int p = 0; p < 4; ++p)
            af.u[g2 * 4 + p] = qS32[m * 16 + half * 4 + g2 * 8 + p];

    v8f sacc[8];
#pragma unroll
    for (int nt = 0; nt < 8; ++nt) {
        int n0 = nt * 16;
#pragma unroll
        for (int i = 0; i < 8; ++i) {   // B: lane=N(seq), K(d) in table order
            int d0 = 8 * half + 16 * (i >> 2) + 2 * (i & 3);
            bf.u[i] = (unsigned)kS[d0 * 128 + n0 + mloc] | ((unsigned)kS[(d0 + 1) * 128 + n0 + mloc] << 16);
        }
        v8f z = {0.f, 0.f, 0.f, 0.f, 0.f, 0.f, 0.f, 0.f};
        sacc[nt] = WMMA_BF16(af.bf, bf.bf, z);
    }

    // --- softmax over 128 columns of each of this lane's 8 rows ---
#pragma unroll
    for (int r = 0; r < 8; ++r) {
        float mx = -3.0e38f;
#pragma unroll
        for (int nt = 0; nt < 8; ++nt) mx = fmaxf(mx, sacc[nt][r]);
#pragma unroll
        for (int off = 1; off < 16; off <<= 1) mx = fmaxf(mx, __shfl_xor(mx, off, 32));
        float s = 0.f;
#pragma unroll
        for (int nt = 0; nt < 8; ++nt) {
            float e = __expf(sacc[nt][r] - mx);
            sacc[nt][r] = e;
            s += e;
        }
#pragma unroll
        for (int off = 1; off < 16; off <<= 1) s += __shfl_xor(s, off, 32);
        float inv = 1.0f / s;
        int row = wave * 16 + r + 8 * half;           // C/D layout: VGPR r -> row
#pragma unroll
        for (int nt = 0; nt < 8; ++nt)
            aS[row * 128 + nt * 16 + mloc] = f2bf(sacc[nt][r] * inv);
    }
    __syncthreads();

    // --- o = attn @ v : K=128 (4 WMMA K-steps), N=32 (2 tiles) ---
    unsigned int* aS32 = (unsigned int*)aS;
    int di = c >> 3, hj = c & 7;
#pragma unroll
    for (int nt = 0; nt < 2; ++nt) {
        v8f acc2 = {0.f, 0.f, 0.f, 0.f, 0.f, 0.f, 0.f, 0.f};
#pragma unroll
        for (int kt = 0; kt < 4; ++kt) {
#pragma unroll
            for (int g2 = 0; g2 < 2; ++g2)
#pragma unroll
                for (int p = 0; p < 4; ++p)
                    af.u[g2 * 4 + p] = aS32[m * 64 + kt * 16 + half * 4 + g2 * 8 + p];
            int kk = kt * 32 + 8 * half;              // B: lane = N(d) col, K(seq) in table order
#pragma unroll
            for (int i = 0; i < 8; ++i) {
                int k0 = kk + 16 * (i >> 2) + 2 * (i & 3);
                bf.u[i] = (unsigned)vS[k0 * 32 + nt * 16 + mloc] | ((unsigned)vS[(k0 + 1) * 32 + nt * 16 + mloc] << 16);
            }
            acc2 = WMMA_BF16(af.bf, bf.bf, acc2);
        }
#pragma unroll
        for (int r = 0; r < 8; ++r) {
            int h = wave * 16 + r + 8 * half;
            int d = nt * 16 + mloc;
            pre[(((b * 16 + di) * 128 + h) * 256) + hj * 32 + d] = f2bf(acc2[r]);
        }
    }
}

// ---------------------------------------------------------------------------
// Kernel 4: out = pre(65536x256 bf16) @ Wt(256x256 bf16) + bias, f32 out.
// One block per 16-row M-tile: A tile (8 KB) async-DMA'd to LDS once, shared
// by 8 waves; each wave owns one 32-col N-group (16 WMMAs).
// ---------------------------------------------------------------------------
__global__ void __launch_bounds__(256) proj_kernel(
    const unsigned short* __restrict__ pre, const unsigned short* __restrict__ wbf,
    const float* __restrict__ bias, float* __restrict__ out)
{
    __shared__ unsigned short aT[16 * 256];              // 8 KB A tile
    int tid   = threadIdx.x;
    int mtile = blockIdx.x;                              // 4096 M-tiles

    unsigned lds_base = (unsigned)(size_t)&aT[0];
    const char* ag = (const char*)(pre + (size_t)mtile * 16 * 256);
#pragma unroll
    for (int k2 = 0; k2 < 2; ++k2) {
        int cid = tid + k2 * 256;                        // 512 b128 chunks
        async_copy_b128(lds_base + (unsigned)(cid << 4), ag + (cid << 4));
    }
    wait_async0();
    __syncthreads();

    int wave = tid >> 5, lane = tid & 31;
    int ng   = wave;                                     // N-group of 32 cols
    int mloc = lane & 15, half = lane >> 4;

    const unsigned int* aT32 = (const unsigned int*)aT;  // 128 dwords per row
    const unsigned int* wg   = (const unsigned int*)wbf; // [k][o], 128 dwords per k
    ABFrag af, bf;

#pragma unroll
    for (int nt = 0; nt < 2; ++nt) {
        int n0 = ng * 32 + nt * 16;
        v8f acc = {0.f, 0.f, 0.f, 0.f, 0.f, 0.f, 0.f, 0.f};
        for (int kt = 0; kt < 8; ++kt) {
#pragma unroll
            for (int g2 = 0; g2 < 2; ++g2)
#pragma unroll
                for (int p = 0; p < 4; ++p)
                    af.u[g2 * 4 + p] = aT32[mloc * 128 + kt * 16 + half * 4 + g2 * 8 + p];
            int kk = kt * 32 + 8 * half;
#pragma unroll
            for (int i = 0; i < 8; ++i) {
                int k0 = kk + 16 * (i >> 2) + 2 * (i & 3);
                bf.u[i] = (unsigned)wbf[k0 * 256 + n0 + mloc] | ((unsigned)wbf[(k0 + 1) * 256 + n0 + mloc] << 16);
            }
            acc = WMMA_BF16(af.bf, bf.bf, acc);
        }
#pragma unroll
        for (int r = 0; r < 8; ++r) {
            int row = mtile * 16 + r + 8 * half;
            int n   = n0 + mloc;
            out[(size_t)row * 256 + n] = acc[r] + bias[n];
        }
    }
}

// ---------------------------------------------------------------------------
extern "C" void kernel_launch(void* const* d_in, const int* in_sizes, int n_in,
                              void* d_out, int out_size, void* d_ws, size_t ws_size,
                              hipStream_t stream)
{
    const float* x  = (const float*)d_in[0];
    const float* w1 = (const float*)d_in[1];
    const float* w2 = (const float*)d_in[2];
    const float* wo = (const float*)d_in[3];
    const float* bo = (const float*)d_in[4];
    float* out = (float*)d_out;

    char* ws = (char*)d_ws;
    const size_t QSZ = (size_t)32 * 128 * 128 * 32 * 2;   // 33.55 MB each (bf16)
    unsigned short* qb  = (unsigned short*)(ws);
    unsigned short* kTb = (unsigned short*)(ws + QSZ);
    unsigned short* vb  = (unsigned short*)(ws + 2 * QSZ);
    unsigned short* pre = (unsigned short*)(ws + 3 * QSZ);
    unsigned short* wbf = (unsigned short*)(ws + 4 * QSZ); // 128 KB

    prep_w_kernel<<<256, 256, 0, stream>>>(wo, wbf);
    conv_qkv_kernel<<<196608, 256, 0, stream>>>(x, w1, w2, qb, kTb, vb);
    attn_kernel<<<4096, 256, 0, stream>>>(qb, kTb, vb, pre);
    proj_kernel<<<4096, 256, 0, stream>>>(pre, wbf, bo, out);
}
